// ManualSVDLinear_49211735277854
// MI455X (gfx1250) — compile-verified, matrix operand order checked by reading, not run on hardware
//
#include <hip/hip_runtime.h>
#include <hip/hip_bf16.h>
#include <stdint.h>

// ---------------------------------------------------------------------------
// ManualSVDLinear forward for MI455X (gfx1250, wave32, WMMA + TDM).
//   out = fake_quant(x*smooth) @ (W + lora_b@lora_a)^T + bias
// M=16384, K=4096, N=4096, R=64, quant block = 128.
// ---------------------------------------------------------------------------

typedef __attribute__((ext_vector_type(16))) _Float16 v16h;
typedef __attribute__((ext_vector_type(8)))  _Float16 v8h;
typedef __attribute__((ext_vector_type(4)))  _Float16 v4h;
typedef __attribute__((ext_vector_type(8)))  float    v8f;

#define GEMM_M 16384
#define GEMM_N 4096
#define GEMM_K 4096
#define RANK   64
#define QBLOCK 128

// ---------------------------------------------------------------------------
// Kernel 1: smooth-scale + block fake-quant (block=128) + convert to f16.
// One wave (32 lanes) per quant block; each lane handles 4 elements.
// ---------------------------------------------------------------------------
__global__ __launch_bounds__(256) void quant_kernel(
    const float* __restrict__ x, const float* __restrict__ smooth,
    _Float16* __restrict__ xq)
{
    const int lane = threadIdx.x & 31;
    const int wave = threadIdx.x >> 5;
    const long long qb = (long long)blockIdx.x * 8 + wave;   // quant-block id
    const long long base = qb * QBLOCK + lane * 4;
    const int col = (int)((qb & 31) * QBLOCK) + lane * 4;    // K = 4096 = 32 blocks/row

    float4 xv = *(const float4*)(x + base);
    float4 sv = *(const float4*)(smooth + col);
    float v0 = xv.x * sv.x, v1 = xv.y * sv.y, v2 = xv.z * sv.z, v3 = xv.w * sv.w;

    float amax = fmaxf(fmaxf(fabsf(v0), fabsf(v1)), fmaxf(fabsf(v2), fabsf(v3)));
    #pragma unroll
    for (int off = 16; off >= 1; off >>= 1)
        amax = fmaxf(amax, __shfl_xor(amax, off, 32));

    float scale = fminf(fmaxf(amax, 1e-12f) * (1.0f / 6.0f), 448.0f);
    float inv = 1.0f / scale;

    float q0 = fminf(fmaxf(rintf(v0 * inv), -6.0f), 6.0f) * scale;
    float q1 = fminf(fmaxf(rintf(v1 * inv), -6.0f), 6.0f) * scale;
    float q2 = fminf(fmaxf(rintf(v2 * inv), -6.0f), 6.0f) * scale;
    float q3 = fminf(fmaxf(rintf(v3 * inv), -6.0f), 6.0f) * scale;

    v4h o;
    o.x = (_Float16)q0; o.y = (_Float16)q1; o.z = (_Float16)q2; o.w = (_Float16)q3;
    *(v4h*)(xq + base) = o;
}

// ---------------------------------------------------------------------------
// Kernel 2: W_eff = W + lora_b @ lora_a, converted to f16.
// ---------------------------------------------------------------------------
__global__ __launch_bounds__(256) void weff_kernel(
    const float* __restrict__ w, const float* __restrict__ la,
    const float* __restrict__ lb, _Float16* __restrict__ weff)
{
    const int d = blockIdx.x * 256 + threadIdx.x;
    const int o = blockIdx.y;
    const float* lbo = lb + (size_t)o * RANK;    // uniform -> s_loads
    float acc = w[(size_t)o * GEMM_K + d];
    #pragma unroll 8
    for (int r = 0; r < RANK; ++r)
        acc = fmaf(lbo[r], la[(size_t)r * GEMM_K + d], acc);
    weff[(size_t)o * GEMM_K + d] = (_Float16)acc;
}

// ---------------------------------------------------------------------------
// Tensor Data Mover: stage a 128-row x 64-half 2D tile into LDS with
// 16-byte row padding (pitch 144 B = 72 halves = 36 banks, conflict-free).
// D# built per CDNA5 ISA.
// ---------------------------------------------------------------------------
#if __has_builtin(__builtin_amdgcn_tensor_load_to_lds)
#define USE_TDM 1
#else
#define USE_TDM 0
#endif

#define TM 128
#define TN 128
#define TK 64
#define LDP 72   // LDS row pitch in halves (64 + 8 pad)

#if USE_TDM
typedef __attribute__((ext_vector_type(4))) unsigned int u32x4;
typedef __attribute__((ext_vector_type(8))) int i32x8;
typedef __attribute__((ext_vector_type(4))) int i32x4;

__device__ __forceinline__ void tdm_load_tile(unsigned lds_addr,
                                              unsigned long long gaddr,
                                              unsigned tensor_rows)
{
    u32x4 g0;
    g0.x = 1u;                                           // count=1, user mode
    g0.y = lds_addr;                                     // lds_addr[63:32 of g0]
    g0.z = (unsigned)gaddr;                              // global_addr[31:0]
    g0.w = (unsigned)((gaddr >> 32) & 0x01ffffffull)     // global_addr[56:32]
         | (2u << 30);                                   // type=2 ("image")
    i32x8 g1;
    // data_size=1 (2B), pad_enable, pad_interval=4 (32 DW), pad_amount=3 (4 DW)
    g1[0] = (int)((1u << 16) | (1u << 20) | (4u << 22) | (3u << 25));
    g1[1] = (int)(((unsigned)GEMM_K & 0xffffu) << 16);   // tensor_dim0 lo16
    g1[2] = (int)(((unsigned)GEMM_K >> 16)               // tensor_dim0 hi16
         | ((tensor_rows & 0xffffu) << 16));             // tensor_dim1 lo16
    g1[3] = (int)((tensor_rows >> 16)                    // tensor_dim1 hi16
         | ((unsigned)TK << 16));                        // tile_dim0 = 64
    g1[4] = (int)TM;                                     // tile_dim1 = 128, tile_dim2 = 0
    g1[5] = (int)GEMM_K;                                 // tensor_dim0_stride lo32
    g1[6] = 0;                                           // stride hi, dim1_stride lo
    g1[7] = 0;
    i32x4 z4 = {0, 0, 0, 0};                             // 2-D tensor: groups 2/3 unused
    i32x8 z8 = {0, 0, 0, 0, 0, 0, 0, 0};
    __builtin_amdgcn_tensor_load_to_lds(g0, g1, z4, z4, z8, 0);
}
#endif

// ---------------------------------------------------------------------------
// Kernel 3: WMMA GEMM  out[M,N] = xq[M,K] * weff[N,K]^T + bias.
// 128x128 workgroup tile, 8 waves in 2(M) x 4(N); wave tile 64x32 = 4x2 WMMA
// tiles. K stage = 64 (two wmma K-steps), double-buffered LDS.
// TDM path: waves 0/1 own the A/B tile DMAs (TENSORcnt); fallback stages
// through VGPRs.  Per the CDNA5 16-bit A layout, lane L (row=L&15, g=L>>4)
// holds K = {8g..8g+7} and {16+8g..+7}: two contiguous b128 LDS reads.
// ---------------------------------------------------------------------------
__global__ __launch_bounds__(256) void gemm_kernel(
    const _Float16* __restrict__ xq, const _Float16* __restrict__ weff,
    const float* __restrict__ bias, float* __restrict__ out)
{
    __shared__ _Float16 As[2][TM * LDP];
    __shared__ _Float16 Bs[2][TM * LDP];

    const int tid  = threadIdx.x;
    const int lane = tid & 31;
    const int wave = tid >> 5;
    const int wm   = wave >> 2;          // 0..1  (64 rows each)
    const int wn   = wave & 3;           // 0..3  (32 cols each)
    const int bm   = blockIdx.y * TM;
    const int bn   = blockIdx.x * TN;

    const int g  = lane >> 4;
    const int rl = lane & 15;
    const int KT = GEMM_K / TK;          // 64 stages

    v8f acc[4][2];
    #pragma unroll
    for (int mt = 0; mt < 4; ++mt)
        #pragma unroll
        for (int nt = 0; nt < 2; ++nt)
            acc[mt][nt] = {};

#if USE_TDM
    const unsigned long long aG =
        (unsigned long long)(uintptr_t)xq + (unsigned long long)bm * GEMM_K * 2;
    const unsigned long long bG =
        (unsigned long long)(uintptr_t)weff + (unsigned long long)bn * GEMM_K * 2;

    if (wave == 0)      tdm_load_tile((unsigned)(uintptr_t)&As[0][0], aG, GEMM_M);
    else if (wave == 1) tdm_load_tile((unsigned)(uintptr_t)&Bs[0][0], bG, GEMM_N);

    for (int kt = 0; kt < KT; ++kt) {
        if (wave < 2) __builtin_amdgcn_s_wait_tensorcnt(0);  // my DMA landed in LDS
        __syncthreads();
        const int cur = kt & 1;

        if (kt + 1 < KT) {               // kick next stage's DMAs
            const unsigned long long koff = (unsigned long long)(kt + 1) * TK * 2;
            const int nxt = cur ^ 1;
            if (wave == 0)
                tdm_load_tile((unsigned)(uintptr_t)&As[nxt][0], aG + koff, GEMM_M);
            else if (wave == 1)
                tdm_load_tile((unsigned)(uintptr_t)&Bs[nxt][0], bG + koff, GEMM_N);
        }

        #pragma unroll
        for (int ks = 0; ks < 2; ++ks) {
            const int kc = ks * 32 + 8 * g;
            v16h af[4];
            #pragma unroll
            for (int mt = 0; mt < 4; ++mt) {
                const _Float16* p = &As[cur][(wm * 64 + mt * 16 + rl) * LDP + kc];
                v8h lo = *(const v8h*)p;
                v8h hi = *(const v8h*)(p + 16);
                af[mt] = __builtin_shufflevector(lo, hi,
                    0,1,2,3,4,5,6,7,8,9,10,11,12,13,14,15);
            }
            v16h bf[2];
            #pragma unroll
            for (int nt = 0; nt < 2; ++nt) {
                const _Float16* p = &Bs[cur][(wn * 32 + nt * 16 + rl) * LDP + kc];
                v8h lo = *(const v8h*)p;
                v8h hi = *(const v8h*)(p + 16);
                bf[nt] = __builtin_shufflevector(lo, hi,
                    0,1,2,3,4,5,6,7,8,9,10,11,12,13,14,15);
            }
            #pragma unroll
            for (int mt = 0; mt < 4; ++mt)
                #pragma unroll
                for (int nt = 0; nt < 2; ++nt)
                    acc[mt][nt] = __builtin_amdgcn_wmma_f32_16x16x32_f16(
                        false, af[mt], false, bf[nt],
                        (short)0, acc[mt][nt], false, false);
        }
    }
#else
    // -------- fallback: synchronous staging through VGPRs ------------------
    // 1024 16-byte chunks per tile, 4 per thread: row = c>>3, col8 = (c&7)*8
    const _Float16* aBase = xq   + (size_t)bm * GEMM_K;
    const _Float16* bBase = weff + (size_t)bn * GEMM_K;
    int rr[4], cc[4];
    #pragma unroll
    for (int i = 0; i < 4; ++i) {
        const int c = tid + 256 * i;
        rr[i] = c >> 3; cc[i] = (c & 7) * 8;
    }
    #pragma unroll
    for (int i = 0; i < 4; ++i) {
        *(uint4*)&As[0][rr[i] * LDP + cc[i]] =
            *(const uint4*)(aBase + (size_t)rr[i] * GEMM_K + cc[i]);
        *(uint4*)&Bs[0][rr[i] * LDP + cc[i]] =
            *(const uint4*)(bBase + (size_t)rr[i] * GEMM_K + cc[i]);
    }
    for (int kt = 0; kt < KT; ++kt) {
        __syncthreads();
        const int cur = kt & 1;
        const bool hasNext = (kt + 1 < KT);
        uint4 na[4], nb[4];
        if (hasNext) {
            const int k0 = (kt + 1) * TK;
            #pragma unroll
            for (int i = 0; i < 4; ++i) {
                na[i] = *(const uint4*)(aBase + (size_t)rr[i] * GEMM_K + k0 + cc[i]);
                nb[i] = *(const uint4*)(bBase + (size_t)rr[i] * GEMM_K + k0 + cc[i]);
            }
        }
        #pragma unroll
        for (int ks = 0; ks < 2; ++ks) {
            const int kc = ks * 32 + 8 * g;
            v16h af[4];
            #pragma unroll
            for (int mt = 0; mt < 4; ++mt) {
                const _Float16* p = &As[cur][(wm * 64 + mt * 16 + rl) * LDP + kc];
                v8h lo = *(const v8h*)p;
                v8h hi = *(const v8h*)(p + 16);
                af[mt] = __builtin_shufflevector(lo, hi,
                    0,1,2,3,4,5,6,7,8,9,10,11,12,13,14,15);
            }
            v16h bf[2];
            #pragma unroll
            for (int nt = 0; nt < 2; ++nt) {
                const _Float16* p = &Bs[cur][(wn * 32 + nt * 16 + rl) * LDP + kc];
                v8h lo = *(const v8h*)p;
                v8h hi = *(const v8h*)(p + 16);
                bf[nt] = __builtin_shufflevector(lo, hi,
                    0,1,2,3,4,5,6,7,8,9,10,11,12,13,14,15);
            }
            #pragma unroll
            for (int mt = 0; mt < 4; ++mt)
                #pragma unroll
                for (int nt = 0; nt < 2; ++nt)
                    acc[mt][nt] = __builtin_amdgcn_wmma_f32_16x16x32_f16(
                        false, af[mt], false, bf[nt],
                        (short)0, acc[mt][nt], false, false);
        }
        if (hasNext) {
            const int nxt = cur ^ 1;
            #pragma unroll
            for (int i = 0; i < 4; ++i) {
                *(uint4*)&As[nxt][rr[i] * LDP + cc[i]] = na[i];
                *(uint4*)&Bs[nxt][rr[i] * LDP + cc[i]] = nb[i];
            }
        }
    }
#endif

    // epilogue: C/D layout -> row = 8*(lane>>4)+r, col = lane&15; add bias
    #pragma unroll
    for (int nt = 0; nt < 2; ++nt) {
        const int colg = bn + wn * 32 + nt * 16 + rl;
        const float bv = bias[colg];
        #pragma unroll
        for (int mt = 0; mt < 4; ++mt) {
            #pragma unroll
            for (int r = 0; r < 8; ++r) {
                const int rowg = bm + wm * 64 + mt * 16 + g * 8 + r;
                out[(size_t)rowg * GEMM_N + colg] = acc[mt][nt][r] + bv;
            }
        }
    }
}

// ---------------------------------------------------------------------------
extern "C" void kernel_launch(void* const* d_in, const int* in_sizes, int n_in,
                              void* d_out, int out_size, void* d_ws, size_t ws_size,
                              hipStream_t stream) {
    (void)in_sizes; (void)n_in; (void)out_size; (void)ws_size;
    const float* x      = (const float*)d_in[0];   // [4,4096,4096]
    const float* w      = (const float*)d_in[1];   // [4096,4096]
    const float* smooth = (const float*)d_in[2];   // [4096]
    const float* la     = (const float*)d_in[3];   // [64,4096]
    const float* lb     = (const float*)d_in[4];   // [4096,64]
    const float* bias   = (const float*)d_in[5];   // [4096]
    float* out = (float*)d_out;

    _Float16* xq   = (_Float16*)d_ws;                                        // 128 MB
    _Float16* weff = (_Float16*)((char*)d_ws + (size_t)GEMM_M * GEMM_K * 2); // 32 MB

    quant_kernel<<<(GEMM_M * (GEMM_K / QBLOCK)) / 8, 256, 0, stream>>>(x, smooth, xq);
    weff_kernel<<<dim3(GEMM_K / 256, GEMM_N), 256, 0, stream>>>(w, la, lb, weff);
    gemm_kernel<<<dim3(GEMM_N / TN, GEMM_M / TM), 256, 0, stream>>>(xq, weff, bias, out);
}
